// Attention_75230647156749
// MI455X (gfx1250) — compile-verified, matrix operand order checked by reading, not run on hardware
//
#include <hip/hip_runtime.h>
#include <hip/hip_bf16.h>
#include <math.h>

// ---------------------------------------------------------------------------
// MI455X (gfx1250) attention. Bandwidth-bound: two (B,H,N,N) fp32 attention
// maps (256MB each) dominate -> single-touch HBM writes, 256KB LDS-resident
// probability strip, f16 WMMA for all GEMMs (60 GFLOP total is free), and the
// post map streamed out with global_store_async_from_lds_b128 overlapping the
// P@V WMMA loop.
// ---------------------------------------------------------------------------

typedef __attribute__((ext_vector_type(16))) _Float16 v16h;
typedef __attribute__((ext_vector_type(4)))  _Float16 v4h;
typedef __attribute__((ext_vector_type(2)))  _Float16 v2h;
typedef __attribute__((ext_vector_type(8)))  float    v8f;
typedef __attribute__((ext_vector_type(4)))  float    f4;
typedef __attribute__((ext_vector_type(2)))  float    f2;

#define BB   4
#define NN   1024
#define DIMM 1024
#define HH   16
#define DHH  64
#define ROWS (BB * NN)

static __device__ __forceinline__ v8f wmma_f16(v16h a, v16h b, v8f c) {
  return __builtin_amdgcn_wmma_f32_16x16x32_f16(false, a, false, b, (short)0, c,
                                                false, false);
}

// Fragment-order position for a 16x32 f16 tile (A: row-per-lane; B: col-per-
// lane, identical formula). Element (rc, k) lives at lane*16 + e where
//   lane = rc + 16*((k&15)>>3),  e = (k&7) + 8*(k>>4)
// Inverse of the ISA 7.12.2 16-bit A/B VGPR layout, so each lane's fragment is
// 16 contiguous f16 (32B) -> ds_load_b128 pairs instead of 16x ds_load_u16.
static __device__ __forceinline__ int frag_pos(int rc, int k) {
  return ((rc + (((k & 15) >> 3) << 4)) << 4) + (k & 7) + ((k >> 4) << 3);
}

// ===========================================================================
// Kernel 1: Y = X @ W (+bias for gate). grid=(ROWS/64, DIM/64, 4), 512 thr.
// ===========================================================================
__global__ void proj_gemm(const float* __restrict__ X,
                          const float* __restrict__ W0, const float* __restrict__ W1,
                          const float* __restrict__ W2, const float* __restrict__ W3,
                          const float* __restrict__ bg,
                          float* __restrict__ Q, float* __restrict__ Kd,
                          float* __restrict__ V, float* __restrict__ G) {
  const float* W; float* Y; bool addb = false;
  switch (blockIdx.z) {
    case 0:  W = W0; Y = Q;  break;
    case 1:  W = W1; Y = Kd; break;
    case 2:  W = W2; Y = V;  break;
    default: W = W3; Y = G;  addb = true; break;
  }
  __shared__ _Float16 Af[4 * 512];   // 4 sub-tiles (16x32) in fragment order
  __shared__ _Float16 Bf[4 * 512];

  const int row0 = blockIdx.x * 64, col0 = blockIdx.y * 64;
  const int w = threadIdx.x >> 5, l = threadIdx.x & 31;
  const int ti = w >> 2, tj = w & 3;
  const int khalf = (l >> 4) * 8;

  // staging coords (fixed per thread)
  const int xr = threadIdx.x >> 3;            // 0..63  (row in X tile)
  const int xk = (threadIdx.x & 7) * 4;       // 0..28  (k quad)
  const int xdst = (xr >> 4) * 512 + frag_pos(xr & 15, xk);

  v8f acc = {};
  for (int k0 = 0; k0 < DIMM; k0 += 32) {
    // X: one float4 per thread -> 4 f16 at consecutive e -> ds_store_b64
    {
      const f4 xv = *(const f4*)&X[(size_t)(row0 + xr) * DIMM + k0 + xk];
      v4h h; h[0]=(_Float16)xv[0]; h[1]=(_Float16)xv[1];
             h[2]=(_Float16)xv[2]; h[3]=(_Float16)xv[3];
      *(v4h*)&Af[xdst] = h;
    }
    // W: two k-pairs per thread -> packed ds_store_b32 each
#pragma unroll
    for (int it = 0; it < 2; ++it) {
      const int p  = threadIdx.x + it * 512;  // 0..1023
      const int c  = p & 63;
      const int kp = (p >> 6) << 1;           // even k
      const float* wp = &W[(size_t)(k0 + kp) * DIMM + col0 + c];
      v2h h; h[0] = (_Float16)wp[0]; h[1] = (_Float16)wp[DIMM];
      *(v2h*)&Bf[(c >> 4) * 512 + frag_pos(c & 15, kp)] = h;
    }
    if (k0 + 32 < DIMM && threadIdx.x < 64)
      __builtin_prefetch(&X[(size_t)(row0 + threadIdx.x) * DIMM + k0 + 32], 0, 1);
    __syncthreads();

    const v16h a  = *(const v16h*)&Af[ti * 512 + l * 16];
    const v16h bf = *(const v16h*)&Bf[tj * 512 + l * 16];
    acc = wmma_f16(a, bf, acc);
    __syncthreads();
  }

  const int col   = col0 + tj * 16 + (l & 15);
  const int rbase = row0 + ti * 16 + khalf;
#pragma unroll
  for (int v = 0; v < 8; ++v) {
    float val = acc[v];
    if (addb) val += bg[col];
    Y[(size_t)(rbase + v) * DIMM + col] = val;
  }
}

// ===========================================================================
// Kernel 2: in-place L2 norm over DH=64 + learned scale, Q and K.
// ===========================================================================
__global__ void qk_norm(float* __restrict__ Q, float* __restrict__ K,
                        const float* __restrict__ qs, const float* __restrict__ ks) {
  const int row = blockIdx.x;
  const int w = threadIdx.x >> 5, l = threadIdx.x & 31;
  for (int pass = 0; pass < 2; ++pass) {
    float* T = pass ? K : Q;
    const float* sc = pass ? ks : qs;
#pragma unroll
    for (int hh = 0; hh < 2; ++hh) {
      const int h = w * 2 + hh;
      float* p = T + (size_t)row * DIMM + h * DHH + l * 2;
      f2 v = *(const f2*)p;
      float ss = v[0] * v[0] + v[1] * v[1];
#pragma unroll
      for (int off = 16; off; off >>= 1) ss += __shfl_xor(ss, off, 32);
      const float inv = 1.0f / fmaxf(sqrtf(ss), 1e-12f);
      f2 o; o[0] = v[0] * inv * sc[l * 2]; o[1] = v[1] * inv * sc[l * 2 + 1];
      *(f2*)p = o;
    }
  }
}

// ===========================================================================
// Kernel 3: attention for one (b,h), 64-query strip. grid=(B*H, N/64), 512 thr.
// 256KB dynamic LDS holds the 64x1024 probability strip; post map is streamed
// LDS->HBM with async stores that overlap the P@V WMMA loop.
// ===========================================================================
__global__ void attn_kernel(const float* __restrict__ Q, const float* __restrict__ K,
                            const float* __restrict__ V,
                            const unsigned char* __restrict__ mask,
                            float* __restrict__ pre, float* __restrict__ post,
                            float* __restrict__ O) {
  extern __shared__ float S[];             // 64 * 1024 floats = 256 KB
  __shared__ float invsum[64];
  __shared__ float red[64][8];

  const int bh = blockIdx.x;
  const int b = bh >> 4, h = bh & 15;
  const int i0 = blockIdx.y * 64;
  const int w = threadIdx.x >> 5, l = threadIdx.x & 31;
  const int qi = w >> 2, sub = w & 3;
  const int khalf = (l >> 4) * 8;

  const float* Qb = Q + (size_t)b * NN * DIMM + h * DHH;
  const float* Kb = K + (size_t)b * NN * DIMM + h * DHH;
  const float* Vb = V + (size_t)b * NN * DIMM + h * DHH;

  // Query A-fragments (16 rows x 64 dims -> two K=32 frags), loaded once.
  v16h aq0, aq1;
  {
    const int qrow = i0 + qi * 16 + (l & 15);
#pragma unroll
    for (int v = 0; v < 8; ++v) {
      int kb = ((v < 4) ? 0 : 16) + khalf + 2 * (v & 3);
      const float* p = Qb + (size_t)qrow * DIMM + kb;
      const f2 q0 = *(const f2*)(p);
      const f2 q1 = *(const f2*)(p + 32);
      aq0[2*v] = (_Float16)q0[0]; aq0[2*v+1] = (_Float16)q0[1];
      aq1[2*v] = (_Float16)q1[0]; aq1[2*v+1] = (_Float16)q1[1];
    }
  }

  // ---- S = 10 * Q K^T over all 64 key tiles (pre map wants unmasked scores)
  for (int t = 0; t < 16; ++t) {
    const int jb = sub + 4 * t;
    const int j  = jb * 16 + (l & 15);
    v16h bk0, bk1;
#pragma unroll
    for (int v = 0; v < 8; ++v) {
      int kb = ((v < 4) ? 0 : 16) + khalf + 2 * (v & 3);
      const float* p = Kb + (size_t)j * DIMM + kb;     // B[d][j] = K[j][d]
      const f2 k0v = *(const f2*)(p);
      const f2 k1v = *(const f2*)(p + 32);
      bk0[2*v] = (_Float16)k0v[0]; bk0[2*v+1] = (_Float16)k0v[1];
      bk1[2*v] = (_Float16)k1v[0]; bk1[2*v+1] = (_Float16)k1v[1];
    }
    v8f acc = {};
    acc = wmma_f16(aq0, bk0, acc);
    acc = wmma_f16(aq1, bk1, acc);

    const bool km = mask[b * NN + j] != 0;
#pragma unroll
    for (int v = 0; v < 8; ++v) {
      const int rloc = qi * 16 + v + khalf;
      const int i = i0 + rloc;
      const float s = acc[v] * 10.0f;
      pre[((size_t)bh * NN + i) * NN + j] = s;         // unmasked pre-softmax
      // scores bounded (~|10.2|) after l2norm+scales -> direct exp is safe
      const float pv = (km && j <= i) ? __expf(s) : 0.0f;
      S[(size_t)rloc * NN + j] = pv;
    }
  }
  __syncthreads();

  // ---- row sums: 8 threads/row, contiguous 512B segments, float4 LDS loads
  {
    const int r = threadIdx.x >> 3, cc = threadIdx.x & 7;
    const f4* sp = (const f4*)&S[(size_t)r * NN + cc * 128];
    float ps = 0.0f;
#pragma unroll 8
    for (int k = 0; k < 32; ++k) { f4 vv = sp[k]; ps += vv[0]+vv[1]+vv[2]+vv[3]; }
    red[r][cc] = ps;
  }
  __syncthreads();
  if ((threadIdx.x & 7) == 0) {
    const int r = threadIdx.x >> 3;
    float s = 0.0f;
#pragma unroll
    for (int k = 0; k < 8; ++k) s += red[r][k];
    invsum[r] = 1.0f / (s + 1e-20f);
  }
  __syncthreads();

  // ---- normalize strip in LDS (float4 in/out)
  {
    f4* S4 = (f4*)S;
    for (int it = 0; it < 32; ++it) {
      const int idx4 = threadIdx.x + it * 512;         // 16384 float4s
      const int r = idx4 >> 8;
      f4 vv = S4[idx4];
      const float is = invsum[r];
      vv[0]*=is; vv[1]*=is; vv[2]*=is; vv[3]*=is;
      S4[idx4] = vv;
    }
  }
  __syncthreads();

  // ---- stream post map: 256KB contiguous LDS -> contiguous HBM, async,
  //      overlapping the P@V WMMA loop below (ASYNCcnt-tracked).
  {
    float* postBase = post + ((size_t)bh * NN + i0) * NN;  // contiguous 64*1024
    const unsigned int ldsBase = (unsigned int)(uintptr_t)&S[0];
    for (int it = 0; it < 32; ++it) {
      const unsigned int off = (threadIdx.x + it * 512) * 16u;  // 16B chunks
      const unsigned int ldsAddr = ldsBase + off;
      const unsigned long long ga = (unsigned long long)(uintptr_t)postBase + off;
      asm volatile("global_store_async_from_lds_b128 %0, %1, off"
                   :: "v"(ga), "v"(ldsAddr) : "memory");
    }
  }

  // ---- O = P @ V, causal chunk skipping (P==0 past the diagonal strip)
  v8f oacc = {};
  const int jmax = i0 + 63;
  for (int c = 0; c < 32; ++c) {
    const int jbase = c * 32;
    if (jbase > jmax) break;
    v16h ap, bv;
#pragma unroll
    for (int v = 0; v < 8; ++v) {
      int kb = ((v < 4) ? 0 : 16) + khalf + 2 * (v & 3);
      const f2 sv = *(const f2*)&S[(size_t)(qi * 16 + (l & 15)) * NN + jbase + kb];
      ap[2*v] = (_Float16)sv[0]; ap[2*v+1] = (_Float16)sv[1];
      const float* vp = Vb + (size_t)(jbase + kb) * DIMM + sub * 16 + (l & 15);
      bv[2*v] = (_Float16)vp[0]; bv[2*v+1] = (_Float16)vp[DIMM];
    }
    oacc = wmma_f16(ap, bv, oacc);
  }
  const int ocol = h * DHH + sub * 16 + (l & 15);
#pragma unroll
  for (int v = 0; v < 8; ++v) {
    const int i = i0 + qi * 16 + v + khalf;
    O[((size_t)b * NN + i) * DIMM + ocol] = oacc[v];
  }

  // drain async LDS->HBM stores before wave exit
  asm volatile("s_wait_asynccnt 0" ::: "memory");
}

// ===========================================================================
// Kernel 4: out = (O_attn * sigmoid(G)) @ Wo, query-masked. proj_gemm tiling.
// ===========================================================================
__global__ void out_gemm(const float* __restrict__ Oa, const float* __restrict__ G,
                         const float* __restrict__ Wo,
                         const unsigned char* __restrict__ mask,
                         float* __restrict__ out) {
  __shared__ _Float16 Af[4 * 512];
  __shared__ _Float16 Bf[4 * 512];

  const int row0 = blockIdx.x * 64, col0 = blockIdx.y * 64;
  const int w = threadIdx.x >> 5, l = threadIdx.x & 31;
  const int ti = w >> 2, tj = w & 3;
  const int khalf = (l >> 4) * 8;

  const int xr = threadIdx.x >> 3;
  const int xk = (threadIdx.x & 7) * 4;
  const int xdst = (xr >> 4) * 512 + frag_pos(xr & 15, xk);

  v8f acc = {};
  for (int k0 = 0; k0 < DIMM; k0 += 32) {
    {
      const size_t off = (size_t)(row0 + xr) * DIMM + k0 + xk;
      const f4 ov = *(const f4*)&Oa[off];
      const f4 gv = *(const f4*)&G[off];
      v4h hh;
#pragma unroll
      for (int e = 0; e < 4; ++e)
        hh[e] = (_Float16)(ov[e] * (1.0f / (1.0f + __expf(-gv[e]))));
      *(v4h*)&Af[xdst] = hh;
    }
#pragma unroll
    for (int it = 0; it < 2; ++it) {
      const int p  = threadIdx.x + it * 512;
      const int c  = p & 63;
      const int kp = (p >> 6) << 1;
      const float* wp = &Wo[(size_t)(k0 + kp) * DIMM + col0 + c];
      v2h hh; hh[0] = (_Float16)wp[0]; hh[1] = (_Float16)wp[DIMM];
      *(v2h*)&Bf[(c >> 4) * 512 + frag_pos(c & 15, kp)] = hh;
    }
    __syncthreads();

    const v16h a  = *(const v16h*)&Af[ti * 512 + l * 16];
    const v16h bf = *(const v16h*)&Bf[tj * 512 + l * 16];
    acc = wmma_f16(a, bf, acc);
    __syncthreads();
  }

  const int col   = col0 + tj * 16 + (l & 15);
  const int rbase = row0 + ti * 16 + khalf;
#pragma unroll
  for (int v = 0; v < 8; ++v) {
    const int grow = rbase + v;                        // = b*N + n
    out[(size_t)grow * DIMM + col] = mask[grow] ? acc[v] : 0.0f;
  }
}

// ===========================================================================
extern "C" void kernel_launch(void* const* d_in, const int* in_sizes, int n_in,
                              void* d_out, int out_size, void* d_ws, size_t ws_size,
                              hipStream_t stream) {
  const float* x   = (const float*)d_in[0];
  const unsigned char* mask = (const unsigned char*)d_in[1];
  const float* Wq  = (const float*)d_in[2];
  const float* Wk  = (const float*)d_in[3];
  const float* Wv  = (const float*)d_in[4];
  const float* qsc = (const float*)d_in[5];
  const float* ksc = (const float*)d_in[6];
  const float* Wg  = (const float*)d_in[7];
  const float* bg  = (const float*)d_in[8];
  const float* Wo  = (const float*)d_in[9];

  float* out  = (float*)d_out;                          // (B,N,DIM)
  float* pre  = out + (size_t)BB * NN * DIMM;           // (B,H,N,N)
  float* post = pre + (size_t)BB * HH * NN * NN;        // (B,H,N,N)

  const size_t M = (size_t)ROWS * DIMM;
  float* Q  = (float*)d_ws;
  float* K  = Q + M;
  float* V  = K + M;
  float* G  = V + M;
  float* Oa = G + M;                                    // 80 MB workspace

  proj_gemm<<<dim3(ROWS / 64, DIMM / 64, 4), 512, 0, stream>>>(
      x, Wq, Wk, Wv, Wg, bg, Q, K, V, G);

  qk_norm<<<ROWS, 256, 0, stream>>>(Q, K, qsc, ksc);

  const size_t smem = (size_t)64 * NN * sizeof(float);  // 256 KB of 320 KB WGP LDS
  (void)hipFuncSetAttribute((const void*)attn_kernel,
                            hipFuncAttributeMaxDynamicSharedMemorySize,
                            (int)smem);
  attn_kernel<<<dim3(BB * HH, NN / 64), 512, smem, stream>>>(
      Q, K, V, mask, pre, post, Oa);

  out_gemm<<<dim3(ROWS / 64, DIMM / 64), 512, 0, stream>>>(
      Oa, G, Wo, mask, out);
}
